// GeneralRNN_34471407517954
// MI455X (gfx1250) — matrix operand order repaired and checked by reading, not compiled, not run on hardware
//
#include <hip/hip_runtime.h>

// ---------------------------------------------------------------------------
// GRU (B=32,T=2048,D=H=256,L=2) for gfx1250 / MI455X.
//   phase 1: xg = X*Wih^T + bih  : bf16 WMMA tile GEMM (M=65536,N=768,K=256)
//   phase 2: sequential scan     : 12 persistent WGs, Whh register-resident,
//            per-step h tile delivered to LDS by the Tensor Data Mover,
//            gate exchange via L2 + atomic barrier (two independent 6-WG groups)
//   round-3 tuning: 2-way split accumulator chains (halved WMMA serial depth,
//            ds_load/WMMA overlap), templated output mode (branch-free update).
// ---------------------------------------------------------------------------

typedef unsigned int        u32;
typedef unsigned short      u16;
typedef unsigned long long  u64;

typedef __attribute__((ext_vector_type(16))) __bf16 v16bf;
typedef __attribute__((ext_vector_type(8)))  float  v8f;
typedef __attribute__((ext_vector_type(4)))  u32    u32x4;
typedef __attribute__((ext_vector_type(8)))  int    i32x8;
typedef __attribute__((ext_vector_type(4)))  int    i32x4;

static constexpr int Bc = 32, Tc = 2048, Dc = 256, Hc = 256, Lc = 2, Gc = 768;

// ---------------- helpers ----------------

__device__ __forceinline__ u16 f2bf(float x) {
  u32 u = __builtin_bit_cast(u32, x);
  u32 r = (u + 0x7FFFu + ((u >> 16) & 1u)) >> 16;   // round-to-nearest-even
  return (u16)r;
}

__device__ __forceinline__ float fsigmoid(float x) {
  return 1.0f / (1.0f + __expf(-x));
}
__device__ __forceinline__ float ftanh(float x) {
  float e = __expf(2.0f * x);
  return 1.0f - 2.0f / (e + 1.0f);                  // saturates correctly at +-1
}

// A fragment (16x32 bf16, M x K) from an LDS tile stored as [16][256] bf16.
__device__ __forceinline__ v16bf load_a_frag_lds(const u32* lds32, int lane, int f) {
  int m   = lane & 15;
  int hi  = lane >> 4;
  int kb2 = f * 16 + hi * 4;                        // dword index of K base
  const u32* row = lds32 + m * 128;                 // 256 bf16 = 128 dwords / row
  union { u32 u[8]; v16bf v; } r;
#pragma unroll
  for (int p = 0; p < 4; ++p) r.u[p]     = row[kb2 + p];
#pragma unroll
  for (int p = 0; p < 4; ++p) r.u[4 + p] = row[kb2 + 8 + p];
  return r.v;
}

// B fragment (32x16 bf16, K x N) with B[k][n] = W[nbase+n][k], W row-major [*][256] bf16.
__device__ __forceinline__ v16bf load_b_frag(const u16* W, int ldk, int nbase, int lane, int f) {
  int n = nbase + (lane & 15);
  int k = f * 32 + (lane >> 4) * 8;
  const u32* p = (const u32*)(W + (size_t)n * ldk + k);
  union { u32 u[8]; v16bf v; } r;
#pragma unroll
  for (int i = 0; i < 8; ++i) r.u[i] = p[i < 4 ? i : i + 4];
  return r.v;
}

// K=256 reduction as two independent WMMA chains (even/odd frags), merged at end.
__device__ __forceinline__ v8f wmma_k256_2chain(const v16bf* afr, const v16bf* bfr) {
  v8f acc0 = {0.f, 0.f, 0.f, 0.f, 0.f, 0.f, 0.f, 0.f};
  v8f acc1 = {0.f, 0.f, 0.f, 0.f, 0.f, 0.f, 0.f, 0.f};
#pragma unroll
  for (int f = 0; f < 8; f += 2) {
    acc0 = __builtin_amdgcn_wmma_f32_16x16x32_bf16(false, afr[f],     false, bfr[f],
                                                   (short)0, acc0, false, false);
    acc1 = __builtin_amdgcn_wmma_f32_16x16x32_bf16(false, afr[f + 1], false, bfr[f + 1],
                                                   (short)0, acc1, false, false);
  }
  return acc0 + acc1;
}

// TDM: async 2D tile load (rows x 256 bf16) global -> LDS. One issue per wave.
__device__ __forceinline__ void tdm_load_2d_bf16(const u16* gsrc, u32 lds_off, u32 rows) {
  u64 ga = (u64)(uintptr_t)gsrc;
  u32x4 g0;
  g0[0] = 1u;                                        // count=1, user descriptor
  g0[1] = lds_off;                                   // LDS byte address
  g0[2] = (u32)(ga & 0xFFFFFFFFu);                   // global_addr[31:0]
  g0[3] = (u32)((ga >> 32) & 0x01FFFFFFu) | (2u << 30); // addr[56:32] | type=2
  i32x8 g1;
  g1[0] = (int)(1u << 16);                           // data_size=1 (2 bytes)
  g1[1] = (int)((256u & 0xFFFFu) << 16);             // tensor_dim0[15:0] @ bits63:48
  g1[2] = (int)(0u | (rows << 16));                  // dim0 hi | tensor_dim1 lo
  g1[3] = (int)(0u | (256u << 16));                  // dim1 hi | tile_dim0=256
  g1[4] = (int)(rows);                               // tile_dim1=rows, tile_dim2=0
  g1[5] = (int)256u;                                 // tensor_dim0_stride lo (elems)
  g1[6] = 0;
  g1[7] = 0;
  i32x4 z4 = {0, 0, 0, 0};
#if defined(__clang_major__) && (__clang_major__ >= 23)
  i32x8 z8 = {0, 0, 0, 0, 0, 0, 0, 0};
  __builtin_amdgcn_tensor_load_to_lds(g0, g1, z4, z4, z8, 0);
#else
  __builtin_amdgcn_tensor_load_to_lds(g0, g1, z4, z4, 0);
#endif
}

// Monotonic 6-WG barrier for one batch-block group (agent scope).
__device__ __forceinline__ void mb_barrier(u32* ctr, u32 expected) {
  __syncthreads();
  if (threadIdx.x == 0) {
    __threadfence();
    atomicAdd(ctr, 1u);
    volatile u32* vc = (volatile u32*)ctr;
    while (*vc < expected) __builtin_amdgcn_s_sleep(1);
    __threadfence();
  }
  __syncthreads();
}

// ---------------- kernel 0: weight conversion + counter reset ----------------

__global__ void prep_kernel(const float* __restrict__ Wih, const float* __restrict__ Whh,
                            u16* __restrict__ Wih_b, u16* __restrict__ Whh_b,
                            u32* __restrict__ ctr) {
  const int n = Lc * Gc * Dc;                        // 393216 per tensor
  for (int i = blockIdx.x * blockDim.x + threadIdx.x; i < n;
       i += gridDim.x * blockDim.x) {
    Wih_b[i] = f2bf(Wih[i]);
    Whh_b[i] = f2bf(Whh[i]);
  }
  if (blockIdx.x == 0 && threadIdx.x < 16) ctr[threadIdx.x] = 0u;
}

// ---------------- kernel 1: xg = X * Wih^T + bih ----------------
// One workgroup (8 waves) per 16-row M tile; A tile staged once in LDS (bf16),
// hoisted into A fragments; each wave covers 6 of the 48 N tiles, K=256.

template <bool IN_F32>
__global__ __launch_bounds__(256) void gate_gemm_kernel(const void* __restrict__ Xv,
                                                        const u16* __restrict__ Wb,
                                                        const float* __restrict__ bih,
                                                        float* __restrict__ xg) {
  __shared__ u16 lds_a[16 * 256];
  const int mt  = blockIdx.x;                        // 0 .. B*T/16-1
  const int tid = threadIdx.x;
  {
    int row  = tid >> 4;
    int colb = (tid & 15) * 16;
    if (IN_F32) {
      const float* src = (const float*)Xv + (size_t)(mt * 16 + row) * Dc + colb;
#pragma unroll
      for (int i = 0; i < 16; ++i) lds_a[row * 256 + colb + i] = f2bf(src[i]);
    } else {
      const u16* src = (const u16*)Xv + (size_t)(mt * 16 + row) * Dc + colb;
#pragma unroll
      for (int i = 0; i < 16; ++i) lds_a[row * 256 + colb + i] = src[i];
    }
  }
  __syncthreads();

  const int wave = tid >> 5, lane = tid & 31;
  const u32* lda = (const u32*)lds_a;
  v16bf afr[8];
#pragma unroll
  for (int f = 0; f < 8; ++f) afr[f] = load_a_frag_lds(lda, lane, f);

#pragma unroll
  for (int j = 0; j < 6; ++j) {
    const int nt = wave + 8 * j;                     // 0..47
    v16bf bfr[8];
#pragma unroll
    for (int f = 0; f < 8; ++f) bfr[f] = load_b_frag(Wb, Dc, nt * 16, lane, f);
    v8f acc = wmma_k256_2chain(afr, bfr);
    const int   col  = nt * 16 + (lane & 15);
    const float bias = bih[col];
    const int   r0   = mt * 16 + (lane >> 4) * 8;
#pragma unroll
    for (int r = 0; r < 8; ++r)
      xg[(size_t)(r0 + r) * Gc + col] = acc[r] + bias;
  }
}

// ---------------- kernel 2: persistent recurrent scan ----------------
// 12 WGs: wg/6 = batch block (rows mb*16..+15), wg%6 = N slice.
// Each wave owns one 16x16 output tile, Whh fragments pinned in VGPRs.
// OUTMODE: 0 -> bf16 sequence (layer-0), 1 -> f32 final output (layer-1).

template <int OUTMODE>
__global__ __launch_bounds__(256) void gru_scan_kernel(const float* __restrict__ xg,
                                                       const u16* __restrict__ Whh_b,
                                                       const float* __restrict__ bhh,
                                                       float* __restrict__ h_f32,
                                                       u16* __restrict__ h_b16,
                                                       float* __restrict__ hg,
                                                       u32* __restrict__ ctr,
                                                       u16* __restrict__ out_b16,
                                                       float* __restrict__ out_f32) {
  __shared__ u16 lds_h[16 * 256];                    // 8 KB h tile (bf16)
  const int wg   = blockIdx.x;                       // 0..11
  const int mb   = wg / 6, sub = wg % 6;
  const int tid  = threadIdx.x;
  const int wave = tid >> 5, lane = tid & 31;
  const int nt   = sub * 8 + wave;                   // 0..47 gate N tile
  u32* mctr = ctr + mb;

  // Whh fragments: register-resident for entire scan (64 VGPRs).
  v16bf bfr[8];
#pragma unroll
  for (int f = 0; f < 8; ++f) bfr[f] = load_b_frag(Whh_b, Hc, nt * 16, lane, f);

  // h0 = 0: zero the LDS tile; updater WG zeroes its fp32 master state.
  {
    u32* l32 = (u32*)lds_h;
#pragma unroll
    for (int i = 0; i < 8; ++i) l32[tid + 256 * i] = 0u;
  }
  if (sub == 0) {
    int row = tid >> 4, colb = (tid & 15) * 16;
    float* hp = h_f32 + (size_t)(mb * 16 + row) * Hc + colb;
#pragma unroll
    for (int i = 0; i < 16; ++i) hp[i] = 0.0f;
  }
  __syncthreads();

  const u32 lds_off = (u32)(uintptr_t)&lds_h[0];
  u32 use = 0;

  for (int t = 0; t < Tc; ++t) {
    if (t > 0) {
      if (wave == 0) {                               // one TDM issue per WG
        tdm_load_2d_bf16(h_b16 + (size_t)mb * 16 * Hc, lds_off, 16u);
        __builtin_amdgcn_s_wait_tensorcnt(0);
      }
      __syncthreads();                               // h tile visible in LDS
    }

    // hg tile = h_tile * Whh_slice^T (two co-executing WMMA chains)
    const u32* l32 = (const u32*)lds_h;
    v16bf afr[8];
#pragma unroll
    for (int f = 0; f < 8; ++f) afr[f] = load_a_frag_lds(l32, lane, f);
    v8f acc = wmma_k256_2chain(afr, bfr);
    {
      const int col = nt * 16 + (lane & 15);
      const int r0  = mb * 16 + (lane >> 4) * 8;
#pragma unroll
      for (int r = 0; r < 8; ++r)
        hg[(size_t)(r0 + r) * Gc + col] = acc[r];
    }

    mb_barrier(mctr, 6u * (++use));                  // all gate slices in L2

    if (sub == 0) {                                  // fused pointwise update
      const int row = tid >> 4, colb = (tid & 15) * 16;
      const int b   = mb * 16 + row;
      const float* hgr = hg + (size_t)b * Gc;
      const float* xgr = xg + ((size_t)b * Tc + t) * Gc;
      float*       hfp = h_f32 + (size_t)b * Hc;

      // pull the xg rows for t+8 toward L2/HBM early (sequential critical path)
      if (t + 8 < Tc) {
        const float* pfr = xg + ((size_t)b * Tc + (t + 8)) * Gc;
        __builtin_prefetch(pfr + colb, 0, 1);
        __builtin_prefetch(pfr + Hc + colb, 0, 1);
        __builtin_prefetch(pfr + 2 * Hc + colb, 0, 1);
      }

#pragma unroll
      for (int i = 0; i < 16; ++i) {
        const int j = colb + i;
        float r  = fsigmoid(xgr[j]       + hgr[j]       + bhh[j]);
        float z  = fsigmoid(xgr[Hc + j]  + hgr[Hc + j]  + bhh[Hc + j]);
        float nn = ftanh(xgr[2*Hc + j] + r * (hgr[2*Hc + j] + bhh[2*Hc + j]));
        float hn = (1.0f - z) * nn + z * hfp[j];
        hfp[j] = hn;
        h_b16[(size_t)b * Hc + j] = f2bf(hn);
        if (OUTMODE == 0) {
          out_b16[((size_t)b * Tc + t) * Hc + j] = f2bf(hn);
        } else {
          __builtin_nontemporal_store(hn, &out_f32[((size_t)b * Tc + t) * Hc + j]);
        }
      }
    }

    mb_barrier(mctr, 6u * (++use));                  // h published before next TDM
  }
}

// ---------------- launcher ----------------

extern "C" void kernel_launch(void* const* d_in, const int* in_sizes, int n_in,
                              void* d_out, int out_size, void* d_ws, size_t ws_size,
                              hipStream_t stream) {
  (void)in_sizes; (void)n_in; (void)out_size; (void)ws_size;
  const float* x   = (const float*)d_in[0];
  const float* Wih = (const float*)d_in[1];
  const float* Whh = (const float*)d_in[2];
  const float* bih = (const float*)d_in[3];
  const float* bhh = (const float*)d_in[4];
  float* out = (float*)d_out;

  // workspace carve-out (~236 MB assumed available)
  char*  ws  = (char*)d_ws;
  size_t off = 0;
  auto carve = [&](size_t bytes) {
    size_t p = off;
    off = (off + bytes + 255) & ~(size_t)255;
    return p;
  };
  float* xg    = (float*)(ws + carve((size_t)Bc * Tc * Gc * 4));  // 192 MB
  u16*   seq_b = (u16*)  (ws + carve((size_t)Bc * Tc * Hc * 2));  //  32 MB
  float* h_f32 = (float*)(ws + carve((size_t)Bc * Hc * 4));
  u16*   h_b16 = (u16*)  (ws + carve((size_t)Bc * Hc * 2));
  float* hg    = (float*)(ws + carve((size_t)Bc * Gc * 4));
  u16*   Wih_b = (u16*)  (ws + carve((size_t)Lc * Gc * Dc * 2));
  u16*   Whh_b = (u16*)  (ws + carve((size_t)Lc * Gc * Hc * 2));
  u32*   ctr   = (u32*)  (ws + carve(64));

  prep_kernel<<<512, 256, 0, stream>>>(Wih, Whh, Wih_b, Whh_b, ctr);

  // layer 0
  gate_gemm_kernel<true><<<(Bc * Tc) / 16, 256, 0, stream>>>(x, Wih_b, bih, xg);
  gru_scan_kernel<0><<<12, 256, 0, stream>>>(xg, Whh_b, bhh, h_f32, h_b16, hg,
                                             ctr + 0, seq_b, nullptr);
  // layer 1
  gate_gemm_kernel<false><<<(Bc * Tc) / 16, 256, 0, stream>>>(
      seq_b, Wih_b + (size_t)Gc * Dc, bih + Gc, xg);
  gru_scan_kernel<1><<<12, 256, 0, stream>>>(xg, Whh_b + (size_t)Gc * Hc, bhh + Gc,
                                             h_f32, h_b16, hg, ctr + 8, nullptr, out);
}
